// RGAT_32804960207442
// MI455X (gfx1250) — matrix operand order repaired and checked by reading, not compiled, not run on hardware
//
#include <hip/hip_runtime.h>
#include <hip/hip_bf16.h>

typedef __attribute__((ext_vector_type(16))) _Float16 v16h;
typedef __attribute__((ext_vector_type(8)))  _Float16 v8h;
typedef __attribute__((ext_vector_type(8)))  float    v8f;

#define DD 64   // embedding dim
#define RR 64   // relation count
#define NEG_SLOPE 0.2f

// ---------- ordered-int float max helpers (monotonic bijection) ----------
__device__ __forceinline__ int f2ord(float f) {
    int i = __float_as_int(f);
    return i >= 0 ? i : (int)(i ^ 0x7FFFFFFF);
}
__device__ __forceinline__ float ord2f(int i) {
    return __int_as_float(i >= 0 ? i : (int)(i ^ 0x7FFFFFFF));
}

// ---------- K0: G = rel_emb @ W^T, stored f16 column-major Bmat[j][d] ----------
// Bmat[j*64+d] = sum_d' W[( (j<64?0:64) + d )*64 + d'] * rel[(j&63)*64 + d']
__global__ void build_B_kernel(const float* __restrict__ W,
                               const float* __restrict__ rel,
                               _Float16* __restrict__ Bmat) {
    int i = blockIdx.x * blockDim.x + threadIdx.x;
    if (i >= 2 * DD * RR) return;
    int j = i >> 6;            // output column 0..127
    int d = i & 63;            // contraction row (emb dim)
    int r = j & 63;
    int wrow = (j < DD ? 0 : DD) + d;
    float s = 0.f;
#pragma unroll 8
    for (int dp = 0; dp < DD; ++dp)
        s += W[wrow * DD + dp] * rel[r * DD + dp];
    Bmat[j * DD + d] = (_Float16)s;
}

// ---------- K1: S[N,128] = emb(f32->f16) @ B  via v_wmma_f32_16x16x32_f16 ----------
// Block: 256 threads = 8 waves. Block owns 16 rows; wave w owns 16-col tile w.
__global__ void score_gemm_kernel(const float* __restrict__ emb,
                                  const _Float16* __restrict__ Bmat,
                                  float* __restrict__ S, int N) {
    __shared__ __align__(32) _Float16 As[16 * DD];        // 2 KB, row-major A tile
    __shared__ __align__(32) _Float16 Bs[128 * DD];       // 16 KB, [col][k] layout
    const int tid  = threadIdx.x;
    const int row0 = blockIdx.x * 16;

    // Load B (already column-major in global): 16384 bytes = 1024 uint4
    {
        const uint4* src = (const uint4*)Bmat;
        uint4*       dst = (uint4*)Bs;
#pragma unroll
        for (int k = 0; k < 4; ++k) dst[tid + 256 * k] = src[tid + 256 * k];
    }
    // Load + convert A tile: 16x64 f32 -> f16 (zero-pad rows >= N)
#pragma unroll
    for (int k = 0; k < 4; ++k) {
        int i   = tid + 256 * k;
        int row = i >> 6, col = i & 63;
        float v = (row0 + row < N) ? emb[(size_t)(row0 + row) * DD + col] : 0.f;
        As[i] = (_Float16)v;
    }
    __syncthreads();

    const int wv   = tid >> 5;        // wave id = column tile
    const int lane = tid & 31;
    const int ln   = lane & 15;       // M (A) / N (B) index
    const int hi   = lane >> 4;       // half-lane group

    v8f acc = {};
#pragma unroll
    for (int h = 0; h < 2; ++h) {     // two K=32 slabs cover K=64
        // A fragment: lane<16 -> K {0..15}+32h ; lane>=16 -> K {16..31}+32h
        v8h alo = *(const v8h*)&As[ln * DD + 32 * h + hi * 16];
        v8h ahi = *(const v8h*)&As[ln * DD + 32 * h + 8 + hi * 16];
        v16h a;
#pragma unroll
        for (int i = 0; i < 8; ++i) { a[i] = alo[i]; a[8 + i] = ahi[i]; }
        // B fragment: column (16*wv+ln), K contiguous thanks to [col][k] LDS layout
        v16h b = *(const v16h*)&Bs[(wv * 16 + ln) * DD + 32 * h + hi * 16];
        acc = __builtin_amdgcn_wmma_f32_16x16x32_f16(
                  false, a, false, b, (short)0, acc, false, false);
    }
    // C/D layout: VGPR j -> row j (lanes 0-15) / row j+8 (lanes 16-31), col = ln
    if (row0 + 16 <= N) {
        // full tile: block-uniform fast path, no per-row exec masking
        float* dst = S + (size_t)(row0 + hi * 8) * 128 + wv * 16 + ln;
#pragma unroll
        for (int j = 0; j < 8; ++j) dst[(size_t)j * 128] = acc[j];
    } else {
#pragma unroll
        for (int j = 0; j < 8; ++j) {
            int row = row0 + j + hi * 8;
            if (row < N) S[(size_t)row * 128 + wv * 16 + ln] = acc[j];
        }
    }
}

// ---------- K2: per-hop init: agg = emb (residual), m = -inf(ord), denom = 0 ----------
__global__ void init_hop_kernel(const float* __restrict__ emb,
                                float* __restrict__ agg,
                                int* __restrict__ mOrd,
                                float* __restrict__ denom, int N) {
    int i = blockIdx.x * blockDim.x + threadIdx.x;
    if (i < N * DD) agg[i] = emb[i];
    if (i < N) { mOrd[i] = (int)0x80000000; denom[i] = 0.f; }
}

// ---------- K3: edge logits + segment max ----------
__global__ void edge_logits_kernel(const int* __restrict__ head,
                                   const int* __restrict__ tail,
                                   const int* __restrict__ etype,
                                   const float* __restrict__ S,
                                   float* __restrict__ ebuf,
                                   int* __restrict__ mOrd, int E) {
    int e = blockIdx.x * blockDim.x + threadIdx.x;
    if (e >= E) return;
    int h = head[e], t = tail[e], r = etype[e];
    float v = S[(size_t)h * 128 + r] + S[(size_t)t * 128 + DD + r];
    v = v > 0.f ? v : NEG_SLOPE * v;                 // leaky_relu
    ebuf[e] = v;
    atomicMax(&mOrd[h], f2ord(v));
}

// ---------- K4: exp(e - max) + segment sum ----------
__global__ void edge_exp_kernel(const int* __restrict__ head,
                                float* __restrict__ ebuf,
                                const int* __restrict__ mOrd,
                                float* __restrict__ denom, int E) {
    int e = blockIdx.x * blockDim.x + threadIdx.x;
    if (e >= E) return;
    int h = head[e];
    float ex = __expf(ebuf[e] - ord2f(mOrd[h]));
    ebuf[e] = ex;
    atomicAdd(&denom[h], ex);
}

// ---------- K5: weighted scatter-sum, one wave32 per edge ----------
__global__ void edge_agg_kernel(const int* __restrict__ head,
                                const int* __restrict__ tail,
                                const float* __restrict__ ebuf,
                                const float* __restrict__ denom,
                                const float* __restrict__ emb,
                                float* __restrict__ agg, int E) {
    int gw   = (blockIdx.x * blockDim.x + threadIdx.x) >> 5;
    int lane = threadIdx.x & 31;
    if (gw >= E) return;
    int h = head[gw], t = tail[gw];
    float alpha = ebuf[gw] / denom[h];
    float v0 = emb[(size_t)t * DD + lane]      * alpha;
    float v1 = emb[(size_t)t * DD + lane + 32] * alpha;
    atomicAdd(&agg[(size_t)h * DD + lane],      v0);
    atomicAdd(&agg[(size_t)h * DD + lane + 32], v1);
}

// ---------- K6: L2 normalize + residual update, one wave32 per node ----------
__global__ void normalize_res_kernel(const float* __restrict__ agg,
                                     float* __restrict__ emb_out,
                                     const float* __restrict__ res_prev,
                                     float* __restrict__ res_out, int N) {
    int gw   = (blockIdx.x * blockDim.x + threadIdx.x) >> 5;
    int lane = threadIdx.x & 31;
    if (gw >= N) return;
    size_t base = (size_t)gw * DD;
    float a0 = agg[base + lane], a1 = agg[base + lane + 32];
    float ss = a0 * a0 + a1 * a1;
#pragma unroll
    for (int off = 16; off; off >>= 1) ss += __shfl_xor(ss, off, 32);
    float inv = 1.f / fmaxf(sqrtf(ss), 1e-12f);
    float e0 = a0 * inv, e1 = a1 * inv;
    emb_out[base + lane]      = e0;
    emb_out[base + lane + 32] = e1;
    res_out[base + lane]      = 0.5f * res_prev[base + lane]      + e0;
    res_out[base + lane + 32] = 0.5f * res_prev[base + lane + 32] + e1;
}

extern "C" void kernel_launch(void* const* d_in, const int* in_sizes, int n_in,
                              void* d_out, int out_size, void* d_ws, size_t ws_size,
                              hipStream_t stream) {
    const int*   edge_index = (const int*)d_in[0];   // [2, E]
    const int*   edge_type  = (const int*)d_in[1];   // [E]
    const float* entity     = (const float*)d_in[2]; // [N, 64]
    const float* rel        = (const float*)d_in[3]; // [64, 64]
    const float* W          = (const float*)d_in[4]; // [128, 64]

    const int E = in_sizes[1];
    const int N = in_sizes[2] / DD;
    float* out = (float*)d_out;

    // ---- workspace carve-up ----
    char* ws = (char*)d_ws;
    float* S      = (float*)ws; ws += (size_t)N * 128 * sizeof(float);
    float* ebuf   = (float*)ws; ws += (size_t)E * sizeof(float);
    int*   mOrd   = (int*)  ws; ws += (size_t)N * sizeof(int);
    float* denom  = (float*)ws; ws += (size_t)N * sizeof(float);
    float* agg    = (float*)ws; ws += (size_t)N * DD * sizeof(float);
    float* embbuf = (float*)ws; ws += (size_t)N * DD * sizeof(float);
    _Float16* Bmat = (_Float16*)ws;

    // G = rel @ W^T (once; constant across hops)
    build_B_kernel<<<(2 * DD * RR + 255) / 256, 256, 0, stream>>>(W, rel, Bmat);

    const float* emb_in   = entity;
    const float* res_prev = entity;
    for (int hop = 0; hop < 2; ++hop) {
        score_gemm_kernel<<<(N + 15) / 16, 256, 0, stream>>>(emb_in, Bmat, S, N);
        init_hop_kernel<<<(N * DD + 255) / 256, 256, 0, stream>>>(emb_in, agg, mOrd, denom, N);
        edge_logits_kernel<<<(E + 255) / 256, 256, 0, stream>>>(
            edge_index, edge_index + E, edge_type, S, ebuf, mOrd, E);
        edge_exp_kernel<<<(E + 255) / 256, 256, 0, stream>>>(
            edge_index, ebuf, mOrd, denom, E);
        edge_agg_kernel<<<(int)(((size_t)E * 32 + 255) / 256), 256, 0, stream>>>(
            edge_index, edge_index + E, ebuf, denom, emb_in, agg, E);
        normalize_res_kernel<<<(int)(((size_t)N * 32 + 255) / 256), 256, 0, stream>>>(
            agg, embbuf, res_prev, out, N);
        emb_in   = embbuf;
        res_prev = out;
    }
}
